// TransformerEncodeLayer_28595892257344
// MI455X (gfx1250) — matrix operand order repaired and checked by reading, not compiled, not run on hardware
//
#include <hip/hip_runtime.h>

// ---------------------------------------------------------------------------
// Types for CDNA5 WMMA (gfx1250, wave32)
// ---------------------------------------------------------------------------
typedef __bf16 bf16_t;
typedef __attribute__((ext_vector_type(16))) __bf16 v16bf;
typedef __attribute__((ext_vector_type(8)))  float  v8f;
typedef __attribute__((ext_vector_type(4)))  float  v4f;
typedef __attribute__((ext_vector_type(4)))  __bf16 v4bf;

// native f32 -> bf16 conversion (lets the compiler use v_cvt*_bf16_f32)
__device__ __forceinline__ bf16_t f2bf(float f) { return (bf16_t)f; }

// 4 f32 -> 4 bf16 packed in uint2 (8 bytes), via vector convert
__device__ __forceinline__ uint2 cvt4(float4 f) {
  v4f x = __builtin_bit_cast(v4f, f);
  v4bf h = __builtin_convertvector(x, v4bf);
  return __builtin_bit_cast(uint2, h);
}
__device__ __forceinline__ uint2 cvt4(uint2 d) { return d; }

// raw 4-element vector load type per source precision
template <typename T> struct Raw4;
template <> struct Raw4<float>  { using t = float4; };
template <> struct Raw4<bf16_t> { using t = uint2; };

// 4 separate bf16 values (for scattered LDS stores)
struct __align__(8) bf4 { bf16_t v[4]; };
__device__ __forceinline__ bf4 tobf4(float4 f) {
  bf4 r; r.v[0] = f2bf(f.x); r.v[1] = f2bf(f.y); r.v[2] = f2bf(f.z); r.v[3] = f2bf(f.w);
  return r;
}
__device__ __forceinline__ bf4 tobf4(uint2 d) { return __builtin_bit_cast(bf4, d); }

__device__ __forceinline__ void store_out(float*  p, float v) { *p = v; }
__device__ __forceinline__ void store_out(bf16_t* p, float v) { *p = f2bf(v); }

// ---------------------------------------------------------------------------
// Tiled bf16 WMMA GEMM:  C = alpha * A(MxK) @ B(KxN) (+ bias) (opt ReLU).
// 256 threads (8 waves), block tile 128(M) x NB(N), K-step 32.
// Wave grid 4(M) x 2(N): each wave computes 32 x NB/2 via 2 x NB/32 WMMAs.
// Batched over blockIdx.z with (b = z>>4, h = z&15) stride pairs so head
// slices of [B,T,H*DK] tensors are addressed directly.
// ---------------------------------------------------------------------------
#define MB 128
#define KB 32
#define AP 36   // LDS pitch (bf16) for A rows: 32 + 4 pad -> 72B, 8B-aligned rows
#define BP 36   // LDS pitch (bf16) for B columns

template <typename TA, typename TB, bool TRANSB, bool RELU, bool HAS_BIAS,
          int NB, typename TC>
__global__ __launch_bounds__(256) void gemm_wmma(
    const TA* __restrict__ Ag, const TB* __restrict__ Bg,
    const float* __restrict__ bias, TC* __restrict__ Cg,
    int M, int N, int K, int lda, int ldb, int ldc,
    long long sAb, long long sAh, long long sBb, long long sBh,
    long long sCb, long long sCh, float alpha)
{
  constexpr int TN  = NB / 32;           // 16-col WMMA tiles per wave
  constexpr int NA4 = (MB * KB) / 1024;  // A 4-elem chunks per thread (4)
  constexpr int NB4 = (KB * NB) / 1024;  // B 4-elem chunks per thread (2 or 4)
  __shared__ __align__(16) bf16_t As[MB * AP];
  __shared__ __align__(16) bf16_t Bs[NB * BP];

  const int tid  = threadIdx.x;
  const int lane = tid & 31;
  const int wave = tid >> 5;
  const int wm   = wave >> 1;            // 0..3  (M direction)
  const int wn   = wave & 1;             // 0..1  (N direction)

  const int z = blockIdx.z;
  const TA* A = Ag + (long long)(z >> 4) * sAb + (long long)(z & 15) * sAh;
  const TB* B = Bg + (long long)(z >> 4) * sBb + (long long)(z & 15) * sBh;
  TC*       C = Cg + (long long)(z >> 4) * sCb + (long long)(z & 15) * sCh;

  const int m0 = blockIdx.y * MB;
  const int n0 = blockIdx.x * NB;

  v8f zero = {};
  v8f acc[2][TN];
#pragma unroll
  for (int i = 0; i < 2; ++i)
#pragma unroll
    for (int j = 0; j < TN; ++j) acc[i][j] = zero;

  const int mr  = lane & 15;
  const int kh2 = (lane >> 4) * 4;   // A: dword offset of K-group (K 0-7 vs 8-15)
  const int kb2 = (lane >> 4) * 8;   // B: dword offset of K-half  (K 0-15 vs 16-31)

  for (int k0 = 0; k0 < K; k0 += KB) {
    // ---- phase 1: issue all global loads for this K tile ----
    typename Raw4<TA>::t ra[NA4];
#pragma unroll
    for (int i = 0; i < NA4; ++i) {
      int idx4 = tid + 256 * i;
      int r = idx4 >> 3;                   // 8 chunks of 4 per 32-wide row
      int c = (idx4 & 7) << 2;
      ra[i] = *(const typename Raw4<TA>::t*)&A[(long long)(m0 + r) * lda + k0 + c];
    }
    typename Raw4<TB>::t rb[NB4];
#pragma unroll
    for (int i = 0; i < NB4; ++i) {
      int idx4 = tid + 256 * i;
      if (TRANSB) {
        int nn = idx4 >> 3, cc = (idx4 & 7) << 2;
        rb[i] = *(const typename Raw4<TB>::t*)&B[(long long)(n0 + nn) * ldb + k0 + cc];
      } else {
        constexpr int N4 = NB / 4;
        int kk = idx4 / N4, nn = (idx4 % N4) << 2;
        rb[i] = *(const typename Raw4<TB>::t*)&B[(long long)(k0 + kk) * ldb + n0 + nn];
      }
    }

    // ---- prefetch next K tile while this one computes ----
    if (k0 + KB < K) {
      int r = tid >> 3, c = (tid & 7) << 2;
      __builtin_prefetch(&A[(long long)(m0 + r) * lda + (k0 + KB) + c], 0, 1);
      if (TRANSB) {
        __builtin_prefetch(&B[(long long)(n0 + r) * ldb + (k0 + KB) + c], 0, 1);
      } else {
        constexpr int N4 = NB / 4;
        __builtin_prefetch(
            &B[(long long)(k0 + KB + tid / N4) * ldb + n0 + ((tid % N4) << 2)], 0, 1);
      }
    }

    // ---- phase 2: convert + stage into LDS ----
#pragma unroll
    for (int i = 0; i < NA4; ++i) {
      int idx4 = tid + 256 * i;
      int r = idx4 >> 3, c = (idx4 & 7) << 2;
      *(uint2*)&As[r * AP + c] = cvt4(ra[i]);
    }
#pragma unroll
    for (int i = 0; i < NB4; ++i) {
      int idx4 = tid + 256 * i;
      if (TRANSB) {
        int nn = idx4 >> 3, cc = (idx4 & 7) << 2;
        *(uint2*)&Bs[nn * BP + cc] = cvt4(rb[i]);
      } else {
        constexpr int N4 = NB / 4;
        int kk = idx4 / N4, nn = (idx4 % N4) << 2;
        bf4 d = tobf4(rb[i]);
#pragma unroll
        for (int j = 0; j < 4; ++j) Bs[(nn + j) * BP + kk] = d.v[j];
      }
    }
    __syncthreads();

    // ---- build fragments per ISA 7.12.2 layouts ----
    union U { v16bf v; unsigned int u[8]; };
    U af[2], fb[TN];
#pragma unroll
    for (int t = 0; t < 2; ++t) {
      // A 16x32: lane<16 row m holds K 0-7 & 16-23; lane>=16 holds K 8-15 & 24-31
      const unsigned int* ar =
          (const unsigned int*)&As[(wm * 32 + t * 16 + mr) * AP];
#pragma unroll
      for (int j = 0; j < 4; ++j) {
        af[t].u[j]     = ar[kh2 + j];
        af[t].u[4 + j] = ar[8 + kh2 + j];
      }
    }
#pragma unroll
    for (int t = 0; t < TN; ++t) {
      // B 32x16: lanes 0-15 col n hold K 0-7 & 16-23; lanes 16-31 hold K 8-15 & 24-31
      const unsigned int* br =
          (const unsigned int*)&Bs[(wn * (NB / 2) + t * 16 + mr) * BP];
#pragma unroll
      for (int j = 0; j < 8; ++j) fb[t].u[j] = br[(j >> 2) * 8 + kh2 + (j & 3)];
    }

#pragma unroll
    for (int tm = 0; tm < 2; ++tm)
#pragma unroll
      for (int tn = 0; tn < TN; ++tn)
        acc[tm][tn] = __builtin_amdgcn_wmma_f32_16x16x32_bf16(
            false, af[tm].v, false, fb[tn].v, (short)0, acc[tm][tn], false, false);

    __syncthreads();
  }

  // ---- epilogue: C/D 16x16 f32 layout (lane = column, vgpr r = row) ----
  const int nloc = lane & 15;
  const int mb8  = (lane >> 4) * 8;
#pragma unroll
  for (int tn = 0; tn < TN; ++tn) {
    const int col = n0 + wn * (NB / 2) + tn * 16 + nloc;
    float bv = 0.0f;
    if (HAS_BIAS) bv = bias[col];
#pragma unroll
    for (int tm = 0; tm < 2; ++tm) {
      TC* cp = C + (long long)(m0 + wm * 32 + tm * 16 + mb8) * ldc + col;
#pragma unroll
      for (int r = 0; r < 8; ++r) {
        float v = acc[tm][tn][r] * alpha + bv;
        if (RELU) v = fmaxf(v, 0.0f);
        store_out(cp, v);
        cp += ldc;
      }
    }
  }
}

// ---------------------------------------------------------------------------
// Row softmax over 2048 columns, with clamp_min(-6000). In place.
// ---------------------------------------------------------------------------
__global__ __launch_bounds__(256) void softmax_rows(float* __restrict__ S)
{
  __shared__ float red[256];
  const int tid = threadIdx.x;
  float* row = S + (long long)blockIdx.x * 2048;

  float v[8];
  float m = -3.4e38f;
#pragma unroll
  for (int i = 0; i < 8; ++i) {
    float x = fmaxf(row[tid + 256 * i], -6000.0f);
    v[i] = x;
    m = fmaxf(m, x);
  }
  red[tid] = m; __syncthreads();
  for (int s = 128; s > 0; s >>= 1) {
    if (tid < s) red[tid] = fmaxf(red[tid], red[tid + s]);
    __syncthreads();
  }
  m = red[0]; __syncthreads();

  float sum = 0.0f;
#pragma unroll
  for (int i = 0; i < 8; ++i) { v[i] = __expf(v[i] - m); sum += v[i]; }
  red[tid] = sum; __syncthreads();
  for (int s = 128; s > 0; s >>= 1) {
    if (tid < s) red[tid] += red[tid + s];
    __syncthreads();
  }
  const float inv = 1.0f / red[0];
#pragma unroll
  for (int i = 0; i < 8; ++i) row[tid + 256 * i] = v[i] * inv;
}

// ---------------------------------------------------------------------------
// Y = alpha * ((X+R) - mu) / (std_unbiased + 1e-6) + beta  per row of 1024.
// ---------------------------------------------------------------------------
__global__ __launch_bounds__(256) void add_norm(
    const float* __restrict__ X, const float* __restrict__ R,
    const float* __restrict__ alpha, const float* __restrict__ beta,
    float* __restrict__ Y)
{
  __shared__ float red[256];
  const int tid = threadIdx.x;
  const long long base = (long long)blockIdx.x * 1024;

  float v[4];
  float s = 0.0f;
#pragma unroll
  for (int i = 0; i < 4; ++i) {
    int c = tid + 256 * i;
    float x = X[base + c] + R[base + c];
    v[i] = x; s += x;
  }
  red[tid] = s; __syncthreads();
  for (int t = 128; t > 0; t >>= 1) {
    if (tid < t) red[tid] += red[tid + t];
    __syncthreads();
  }
  const float mu = red[0] * (1.0f / 1024.0f);
  __syncthreads();

  float d2 = 0.0f;
#pragma unroll
  for (int i = 0; i < 4; ++i) { float d = v[i] - mu; d2 += d * d; }
  red[tid] = d2; __syncthreads();
  for (int t = 128; t > 0; t >>= 1) {
    if (tid < t) red[tid] += red[tid + t];
    __syncthreads();
  }
  const float sd = sqrtf(red[0] * (1.0f / 1023.0f));   // ddof=1 like torch
  const float inv = 1.0f / (sd + 1e-6f);
#pragma unroll
  for (int i = 0; i < 4; ++i) {
    int c = tid + 256 * i;
    Y[base + c] = alpha[c] * ((v[i] - mu) * inv) + beta[c];
  }
}

// ---------------------------------------------------------------------------
// Launcher: full encoder layer pipeline.
// ---------------------------------------------------------------------------
extern "C" void kernel_launch(void* const* d_in, const int* in_sizes, int n_in,
                              void* d_out, int out_size, void* d_ws, size_t ws_size,
                              hipStream_t stream)
{
  (void)in_sizes; (void)n_in; (void)out_size; (void)ws_size;

  constexpr int Bz = 2, T = 2048, D = 1024, H = 16, DK = 64, FF = 4096;
  constexpr int BT = Bz * T;           // 4096 token rows
  constexpr long long TD  = (long long)T * D;
  constexpr long long TT  = (long long)T * T;
  constexpr long long HTT = (long long)H * TT;

  const float* x  = (const float*)d_in[0];
  // d_in[1] = x_mask (unused by reference forward)
  const float* Wq = (const float*)d_in[2];  const float* bq = (const float*)d_in[3];
  const float* Wk = (const float*)d_in[4];  const float* bk = (const float*)d_in[5];
  const float* Wv = (const float*)d_in[6];  const float* bv = (const float*)d_in[7];
  const float* Wo = (const float*)d_in[8];  const float* bo = (const float*)d_in[9];
  const float* W1 = (const float*)d_in[10]; const float* b1 = (const float*)d_in[11];
  const float* W2 = (const float*)d_in[12]; const float* b2 = (const float*)d_in[13];
  const float* a1 = (const float*)d_in[14]; const float* be1 = (const float*)d_in[15];
  const float* a2 = (const float*)d_in[16]; const float* be2 = (const float*)d_in[17];

  float* out   = (float*)d_out;                       // [B,T,D]
  float* score = out + (size_t)BT * D;                // [B,H,T,T] (second output)

  char* w = (char*)d_ws;
  bf16_t* qb   = (bf16_t*)w; w += (size_t)BT * D * sizeof(bf16_t);
  bf16_t* kb   = (bf16_t*)w; w += (size_t)BT * D * sizeof(bf16_t);
  bf16_t* vb   = (bf16_t*)w; w += (size_t)BT * D * sizeof(bf16_t);
  float*  val  = (float*)w;  w += (size_t)BT * D * sizeof(float);
  float*  attn = (float*)w;  w += (size_t)BT * D * sizeof(float);
  float*  xn   = (float*)w;  w += (size_t)BT * D * sizeof(float);
  float*  ffo  = (float*)w;  w += (size_t)BT * D * sizeof(float);
  bf16_t* hid  = (bf16_t*)w; w += (size_t)BT * FF * sizeof(bf16_t);

  const dim3 blk(256);

  // ---- Q/K/V projections: x[4096,1024] @ W[1024,1024] + b -> bf16 ----
  const dim3 gproj(D / 128, BT / MB, 1);   // (8, 32)
  gemm_wmma<float, float, false, false, true, 128, bf16_t><<<gproj, blk, 0, stream>>>(
      x, Wq, bq, qb, BT, D, D, D, D, D, 0, 0, 0, 0, 0, 0, 1.0f);
  gemm_wmma<float, float, false, false, true, 128, bf16_t><<<gproj, blk, 0, stream>>>(
      x, Wk, bk, kb, BT, D, D, D, D, D, 0, 0, 0, 0, 0, 0, 1.0f);
  gemm_wmma<float, float, false, false, true, 128, bf16_t><<<gproj, blk, 0, stream>>>(
      x, Wv, bv, vb, BT, D, D, D, D, D, 0, 0, 0, 0, 0, 0, 1.0f);

  // ---- scores: S[b,h] = (Q_h @ K_h^T) / sqrt(64), batched over 32 heads ----
  const dim3 gsc(T / 128, T / MB, Bz * H); // (16, 16, 32)
  gemm_wmma<bf16_t, bf16_t, true, false, false, 128, float><<<gsc, blk, 0, stream>>>(
      qb, kb, nullptr, score, T, T, DK, D, D, T,
      TD, DK, TD, DK, HTT, TT, 0.125f);

  // ---- clamp + softmax in place (this buffer IS the score output) ----
  softmax_rows<<<dim3(Bz * H * T), blk, 0, stream>>>(score);

  // ---- val[b,h] = P @ V_h  (M=2048, N=64, K=2048), head-interleaved out ----
  const dim3 gav(1, T / MB, Bz * H);       // (1, 16, 32)
  gemm_wmma<float, bf16_t, false, false, false, 64, float><<<gav, blk, 0, stream>>>(
      score, vb, nullptr, val, T, DK, T, T, D, D,
      HTT, TT, TD, DK, TD, DK, 1.0f);

  // ---- output projection ----
  gemm_wmma<float, float, false, false, true, 128, float><<<gproj, blk, 0, stream>>>(
      val, Wo, bo, attn, BT, D, D, D, D, D, 0, 0, 0, 0, 0, 0, 1.0f);

  // ---- residual + norm1 ----
  add_norm<<<dim3(BT), blk, 0, stream>>>(x, attn, a1, be1, xn);

  // ---- FFN: relu(xn @ W1 + b1) -> bf16 hidden; hidden @ W2 + b2 ----
  const dim3 gff1(FF / 128, BT / MB, 1);   // (32, 32)
  gemm_wmma<float, float, false, true, true, 128, bf16_t><<<gff1, blk, 0, stream>>>(
      xn, W1, b1, hid, BT, FF, D, D, FF, FF, 0, 0, 0, 0, 0, 0, 1.0f);
  const dim3 gff2(D / 128, BT / MB, 1);    // (8, 32)
  gemm_wmma<bf16_t, float, false, false, true, 128, float><<<gff2, blk, 0, stream>>>(
      hid, W2, b2, ffo, BT, D, FF, FF, D, D, 0, 0, 0, 0, 0, 0, 1.0f);

  // ---- residual + norm2 -> primary output ----
  add_norm<<<dim3(BT), blk, 0, stream>>>(xn, ffo, a2, be2, out);
}
